// SPHConvNet_24043226923470
// MI455X (gfx1250) — matrix operand order repaired and checked by reading, not compiled, not run on hardware
//
#include <hip/hip_runtime.h>
#include <hip/hip_bf16.h>

typedef __attribute__((ext_vector_type(16))) _Float16 v16h;
typedef __attribute__((ext_vector_type(8)))  _Float16 v8h;
typedef __attribute__((ext_vector_type(8)))  float    v8f;
typedef __attribute__((ext_vector_type(2)))  float    v2f;
typedef __attribute__((ext_vector_type(4)))  unsigned int v4u;
typedef __attribute__((ext_vector_type(8)))  int      v8i;
typedef __attribute__((ext_vector_type(4)))  int      v4i;

#define B_DIM 2
#define N_DIM 4096
#define C_INV 16
#define C_OUTV 32
#define KNN  64
#define NRAD 4
#define PSTR 72            // padded LDS row stride in halves (144 B, 16-B aligned)
#define FLT_BIG 3.0e38f

// ---------------------------------------------------------------------------
// Kernel 1: pairwise-distance tiles via v_wmma_f32_16x16x4_f32 + top-64 select
// block = 512 threads (16 waves); each block owns 16 rows; dist slab in LDS
// ---------------------------------------------------------------------------
__global__ __launch_bounds__(512) void knn_kernel(const float* __restrict__ xyz,
                                                  int* __restrict__ knn_idx,
                                                  float* __restrict__ knn_d2) {
    extern __shared__ char smem[];
    float* dist   = (float*)smem;            // [16][4096]  256 KB
    float* rowdat = dist + 16 * N_DIM;       // [16][4]  x,y,z,r2

    const int tid  = threadIdx.x;
    const int lane = tid & 31;
    const int wave = tid >> 5;
    const int b    = blockIdx.x >> 8;        // 256 row-tiles per batch
    const int tile = blockIdx.x & 255;
    const int row0 = tile * 16;

    if (tid < 16) {
        const float* p = xyz + ((size_t)b * N_DIM + row0 + tid) * 3;
        float x = p[0], y = p[1], z = p[2];
        rowdat[tid * 4 + 0] = x;
        rowdat[tid * 4 + 1] = y;
        rowdat[tid * 4 + 2] = z;
        rowdat[tid * 4 + 3] = x * x + y * y + z * z;
    }
    __syncthreads();

    const int hi = lane >> 4;
    const int nl = lane & 15;

    // A fragment (16x4, f32): lanes 0-15 hold K=0,1 ; lanes 16-31 hold K=2,3.
    // Fold the -2 of (r2_i - 2*dot + r2_j) into A.
    float rx = rowdat[nl * 4 + 0];
    float ry = rowdat[nl * 4 + 1];
    float rz = rowdat[nl * 4 + 2];
    v2f afrag;
    afrag.x = hi ? (-2.0f * rz) : (-2.0f * rx);
    afrag.y = hi ? 0.0f : (-2.0f * ry);

    // 16 waves x 16 cols per iter x 16 iters = 4096 columns
    for (int it = 0; it < 16; ++it) {
        int col = it * 256 + wave * 16 + nl;
        const float* p = xyz + ((size_t)b * N_DIM + col) * 3;
        float cx = p[0], cy = p[1], cz = p[2];
        float r2c = cx * cx + cy * cy + cz * cz;
        v2f bfrag;                            // B (4x16): same lane split as A
        bfrag.x = hi ? cz : cx;
        bfrag.y = hi ? 0.0f : cy;
        v8f acc;                              // C[m,n] = r2_row[m] + r2_col[n]
#pragma unroll
        for (int j = 0; j < 8; ++j) acc[j] = rowdat[(j + 8 * hi) * 4 + 3] + r2c;
        v8f d = __builtin_amdgcn_wmma_f32_16x16x4_f32(false, afrag, false, bfrag,
                                                      (short)0, acc, false, false);
#pragma unroll
        for (int j = 0; j < 8; ++j)
            dist[(j + 8 * hi) * N_DIM + col] = d[j];
    }
    __syncthreads();

    // Top-64 selection: wave w handles row w via shuffle-butterfly argmin.
    {
        float* drow = dist + wave * N_DIM;
        float lmin = FLT_BIG;
        int   lidx = 0;
        for (int i = lane; i < N_DIM; i += 32) {
            float v = drow[i];
            if (v < lmin) { lmin = v; lidx = i; }
        }
        size_t outBase = ((size_t)b * N_DIM + row0 + wave) * KNN;
        for (int r = 0; r < KNN; ++r) {
            float bv = lmin;
            int   bi = lidx;
#pragma unroll
            for (int off = 16; off > 0; off >>= 1) {
                float ov = __shfl_xor(bv, off, 32);
                int   oi = __shfl_xor(bi, off, 32);
                if (ov < bv || (ov == bv && oi < bi)) { bv = ov; bi = oi; }
            }
            if (lane == 0) {
                knn_idx[outBase + r] = bi;
                knn_d2[outBase + r]  = bv;
            }
            if ((bi & 31) == lane) {          // owner invalidates + rescans
                drow[bi] = FLT_BIG;
                lmin = FLT_BIG; lidx = 0;
                for (int i = lane; i < N_DIM; i += 32) {
                    float v = drow[i];
                    if (v < lmin) { lmin = v; lidx = i; }
                }
            }
        }
    }
}

// ---------------------------------------------------------------------------
// Kernel 2: per-point SH conv kernel build + 64x16x64 GEMM via
// v_wmma_f32_16x16x32_f16 (LDS staged in fragment-native transposed layout,
// loaded as b128), weight tensor pulled in by the Tensor Data Mover.
// block = 64 threads (2 waves); one block per point.
// ---------------------------------------------------------------------------
__global__ __launch_bounds__(64) void conv_kernel(const float* __restrict__ xyz,
                                                  const float* __restrict__ signal,
                                                  const float* __restrict__ weight,
                                                  const float* __restrict__ biases,
                                                  const int* __restrict__ knn_idx,
                                                  const float* __restrict__ knn_d2,
                                                  float* __restrict__ out) {
    __shared__ __align__(16) _Float16 ckT[64 * PSTR];   // [rn][p] A matrix, K-contiguous
    __shared__ __align__(16) _Float16 sgT[C_INV * PSTR];// [c][p]  B matrix, K-contiguous
    __shared__ float    tmat[64 * C_INV];               // [rn][c] GEMM result
    __shared__ float    tval[C_INV * NRAD * 4];         // [c][r][l]
    __shared__ float    red[KNN];
    __shared__ float    ywInv;
    __shared__ __align__(16) float wlds[C_OUTV * C_INV * NRAD * 4]; // 8 KB weights via TDM

    const int tid = threadIdx.x;
    const int bv  = blockIdx.x;              // b*N + v
    const int b   = bv >> 12;

    // --- Tensor Data Mover: DMA the 8 KB weight tensor into LDS ----------
    // D# group0/group1 per CDNA5 ISA 8.3/8.4: 1D tile of 2048 x 4-byte elems.
    if (tid < 32) {                          // wave 0 issues; EXEC is ignored by TDM
        unsigned long long ga = (unsigned long long)(const void*)weight;
        unsigned int lds_a   = (unsigned int)(unsigned long long)(const void*)wlds;
        v4u g0;
        g0.x = 1u;                            // count=1, is_restore=0, gather off
        g0.y = lds_a;                         // lds_addr
        g0.z = (unsigned int)ga;              // global_addr[31:0]
        g0.w = (unsigned int)((ga >> 32) & 0x01FFFFFFu) | (2u << 30); // addr[56:32] | type=2
        v8i g1;
        g1[0] = (int)(2u << 16);              // wg_mask=0, data_size=2 (4B)
        g1[1] = (int)(2048u << 16);           // tensor_dim0 lo16 << 16
        g1[2] = (int)(1u << 16);              // tensor_dim0 hi16=0 | tensor_dim1=1
        g1[3] = (int)(2048u << 16);           // tile_dim0 = 2048
        g1[4] = 1;                            // tile_dim1 = 1, tile_dim2 = 0
        g1[5] = 2048;                         // tensor_dim0_stride lo32
        g1[6] = 0;                            // stride0 hi | stride1 lo16<<16
        g1[7] = 0;
        v4i gz4 = {0, 0, 0, 0};
        v8i gz8 = {0, 0, 0, 0, 0, 0, 0, 0};
        __builtin_amdgcn_tensor_load_to_lds(g0, g1, gz4, gz4, gz8, 0);
    }

    // --- per-patch geometry (thread t = patch p) -------------------------
    const int p   = tid;
    const int nbr = knn_idx[(size_t)bv * KNN + p];
    const float* sp = signal + ((size_t)b * N_DIM + nbr) * C_INV;
    __builtin_prefetch(sp, 0, 1);            // global_prefetch_b8: hide gather latency

    const float* cp = xyz + (size_t)bv * 3;
    const float* np = xyz + ((size_t)b * N_DIM + nbr) * 3;
    float dx = np[0] - cp[0];
    float dy = np[1] - cp[1];
    float dz = np[2] - cp[2];
    float pn  = sqrtf(fmaxf(dx * dx + dy * dy + dz * dz, 1e-12f));
    float inv = 1.0f / pn;
    float x = dx * inv, y = dy * inv, z = dz * inv;

    float Y[16];
    Y[0]  = 0.28209479177387814f;
    Y[1]  = 0.4886025119029199f * y;
    Y[2]  = 0.4886025119029199f * z;
    Y[3]  = 0.4886025119029199f * x;
    Y[4]  = 1.0925484305920792f * x * y;
    Y[5]  = 1.0925484305920792f * y * z;
    Y[6]  = 0.31539156525252005f * (3.0f * z * z - 1.0f);
    Y[7]  = 1.0925484305920792f * x * z;
    Y[8]  = 0.5462742152960396f * (x * x - y * y);
    Y[9]  = 0.5900435899266435f * y * (3.0f * x * x - y * y);
    Y[10] = 2.890611442640554f * x * y * z;
    Y[11] = 0.4570457994644658f * y * (5.0f * z * z - 1.0f);
    Y[12] = 0.3731763325901154f * z * (5.0f * z * z - 3.0f);
    Y[13] = 0.4570457994644658f * x * (5.0f * z * z - 1.0f);
    Y[14] = 1.445305721320277f * z * (x * x - y * y);
    Y[15] = 0.5900435899266435f * x * (x * x - y * y);

    float dd = sqrtf(fmaxf(knn_d2[(size_t)bv * KNN + p], 1e-4f));
    float radial[NRAD];
#pragma unroll
    for (int r = 0; r < NRAD; ++r) {
        float rd = dd - 0.1f * r;            // rgrid = {0,.1,.2,.3}, sigma=.1
        radial[r] = __expf(-rd * rd * 50.0f);
    }
    red[p] = radial[0];

#pragma unroll
    for (int c = 0; c < C_INV; ++c) sgT[c * PSTR + p] = (_Float16)sp[c];
    __syncthreads();

    if (tid == 0) {
        float s = 0.0f;
        for (int i = 0; i < KNN; ++i) s += red[i];
        ywInv = 1.0f / (0.28209479177387814f * s + 1e-6f);
    }
    __syncthreads();

    float sc = ywInv;
#pragma unroll
    for (int r = 0; r < NRAD; ++r)
#pragma unroll
        for (int n = 0; n < 16; ++n)
            ckT[(r * 16 + n) * PSTR + p] = (_Float16)(Y[n] * radial[r] * sc);
    __syncthreads();

    // --- GEMM: t[rn, c] = sum_p ck[p, rn] * sg[p, c] ---------------------
    const int wave = tid >> 5;
    const int lane = tid & 31;
    const int hi   = lane >> 4;
    const int nl   = lane & 15;

    // B fragments (shared by all M-tiles): lane holds K = kk + hi*16 .. +15
    v16h bfr[2];
#pragma unroll
    for (int kc = 0; kc < 2; ++kc) {
        const v8h* brow = (const v8h*)(sgT + nl * PSTR + kc * 32 + hi * 16);
        v8h b0 = brow[0];
        v8h b1 = brow[1];
        bfr[kc] = __builtin_shufflevector(b0, b1, 0, 1, 2, 3, 4, 5, 6, 7,
                                          8, 9, 10, 11, 12, 13, 14, 15);
    }

    for (int t8 = 0; t8 < 2; ++t8) {
        int m0 = (wave * 2 + t8) * 16;        // rn-tile base (0,16,32,48)
        v8f acc = {};
#pragma unroll
        for (int kc = 0; kc < 2; ++kc) {
            // A fragment: lane m = m0+nl, K = kk + hi*16 .. +15
            const _Float16* arow = ckT + (m0 + nl) * PSTR + kc * 32 + hi * 16;
            v8h a0 = *(const v8h*)(arow);
            v8h a1 = *(const v8h*)(arow + 8);
            v16h a = __builtin_shufflevector(a0, a1, 0, 1, 2, 3, 4, 5, 6, 7,
                                             8, 9, 10, 11, 12, 13, 14, 15);
            acc = __builtin_amdgcn_wmma_f32_16x16x32_f16(false, a, false, bfr[kc],
                                                         (short)0, acc, false, false);
        }
#pragma unroll
        for (int j = 0; j < 8; ++j)
            tmat[(m0 + j + 8 * hi) * C_INV + nl] = acc[j];
    }

    if (tid < 32) __builtin_amdgcn_s_wait_tensorcnt(0); // weights landed in LDS
    __syncthreads();

    // --- per-l energy: tval[c,r,l] = sqrt(max(sum_n t^2, 1e-4)) ----------
#pragma unroll
    for (int q = 0; q < 4; ++q) {
        int id = tid * 4 + q;                 // id = c*16 + r*4 + l
        int c = id >> 4;
        int r = (id >> 2) & 3;
        int l = id & 3;
        float s = 0.0f;
        for (int n = l * l; n < (l + 1) * (l + 1); ++n) {
            float tv = tmat[(r * 16 + n) * C_INV + c];
            s += tv * tv;
        }
        tval[id] = sqrtf(fmaxf(s, 1e-4f));
    }
    __syncthreads();

    // --- out[i] = bias[i] + sum_{c,r,l} W[i,c,r,l] * tval[c,r,l] ---------
    if (tid < C_OUTV) {
        float accum = biases[tid];
        const float* w = wlds + (size_t)tid * C_INV * NRAD * 4;
        for (int c = 0; c < C_INV; ++c)
            for (int r = 0; r < NRAD; ++r)
#pragma unroll
                for (int l = 0; l < 4; ++l)
                    accum += w[(c * NRAD + r) * 4 + l] * tval[(c * NRAD + r) * 4 + l];
        out[(size_t)bv * C_OUTV + tid] = accum;
    }
}

// ---------------------------------------------------------------------------
extern "C" void kernel_launch(void* const* d_in, const int* in_sizes, int n_in,
                              void* d_out, int out_size, void* d_ws, size_t ws_size,
                              hipStream_t stream) {
    const float* xyz    = (const float*)d_in[0];
    const float* signal = (const float*)d_in[1];
    const float* weight = (const float*)d_in[2];
    const float* biases = (const float*)d_in[3];
    float* out = (float*)d_out;

    int*   knn_idx = (int*)d_ws;
    float* knn_d2  = (float*)((char*)d_ws + (size_t)B_DIM * N_DIM * KNN * sizeof(int));

    size_t shmem = (size_t)(16 * N_DIM + 16 * 4) * sizeof(float);  // ~257 KB of 320 KB WGP LDS
    knn_kernel<<<dim3(B_DIM * (N_DIM / 16)), dim3(512), shmem, stream>>>(xyz, knn_idx, knn_d2);
    conv_kernel<<<dim3(B_DIM * N_DIM), dim3(64), 0, stream>>>(xyz, signal, weight, biases,
                                                              knn_idx, knn_d2, out);
}
